// RNNStepPredictor_30193620091215
// MI455X (gfx1250) — compile-verified, matrix operand order checked by reading, not run on hardware
//
#include <hip/hip_runtime.h>
#include <hip/hip_bf16.h>
#include <math.h>
#include <stdint.h>

// ---------------- problem constants (from reference) ----------------
#define BB     32
#define SS     8
#define CC     16
#define TT     64
#define WUL    16
#define TRL    48                 // TT - WUL
#define NSEQ   4096               // BB*SS*CC
#define ULSTM  128
#define ZW     512                // 4*ULSTM
#define FCN    256
#define CONCAT_K 14336            // WUL*ULSTM + TRL*FCN
#define OSZ    (NSEQ * TRL * 2)   // elements per output slice = 393216

// ---------------- WMMA fragment types ----------------
typedef _Float16 v16h __attribute__((ext_vector_type(16)));
typedef _Float16 v8h  __attribute__((ext_vector_type(8)));
typedef float    v8f  __attribute__((ext_vector_type(8)));

// Fragment loader for 16x16x32 f16 WMMA (A from act tile [16][K], B from
// transposed weight [N][K]).  ISA layout: lanes 0-15 hold K {0..7,16..23},
// lanes 16-31 hold K {8..15,24..31} of its row/col -> two contiguous 16B loads.
__device__ __forceinline__ v16h frag_ld(const _Float16* __restrict__ base,
                                        int rc, int stride, int kbase, int lane) {
    const int kh = kbase + ((lane & 16) ? 8 : 0);
    const _Float16* p = base + (size_t)rc * stride + kh;
    v8h lo = *reinterpret_cast<const v8h*>(p);
    v8h hi = *reinterpret_cast<const v8h*>(p + 16);
    v16h f;
#pragma unroll
    for (int i = 0; i < 8; ++i) { f[i] = lo[i]; f[i + 8] = hi[i]; }
    return f;
}

__device__ __forceinline__ float sigf(float x) { return 1.0f / (1.0f + __expf(-x)); }

// async global -> LDS 16B copy (CDNA5 GLOBAL_LOAD_ASYNC_TO_LDS_B128, ASYNCcnt)
__device__ __forceinline__ void async_cp16(uint32_t lds_off, const _Float16* g) {
    asm volatile("global_load_async_to_lds_b128 %0, %1, off"
                 :: "v"(lds_off), "v"((uint64_t)(uintptr_t)g) : "memory");
}

// ---------------- weight convert + transpose: dst[n][k] = (f16)src[k][n] ----
__global__ void convT_kernel(const float* __restrict__ src, _Float16* __restrict__ dst,
                             int K, int N) {
    int total = K * N;
    for (int i = blockIdx.x * blockDim.x + threadIdx.x; i < total;
         i += gridDim.x * blockDim.x) {
        int k = i / N, n = i - k * N;
        dst[(size_t)n * K + k] = (_Float16)src[(size_t)k * N + n];
    }
}

__global__ void zero_kernel(float* __restrict__ p, int n) {
    for (int i = blockIdx.x * blockDim.x + threadIdx.x; i < n;
         i += gridDim.x * blockDim.x) p[i] = 0.0f;
}

// out[3] = inputs[:,:,:,WUL:,:]
__global__ void copy3_kernel(const float* __restrict__ x, float* __restrict__ dst) {
    int total = NSEQ * TRL * 2;
    for (int i = blockIdx.x * blockDim.x + threadIdx.x; i < total;
         i += gridDim.x * blockDim.x) {
        int n = i / (TRL * 2), j = i - n * (TRL * 2);
        dst[i] = x[(size_t)n * (TT * 2) + WUL * 2 + j];
    }
}

// ---------------- LSTM: 16 sequences per WG, 8 wave32s -----------------------
__global__ __launch_bounds__(256)
void lstm_kernel(const float* __restrict__ x,       // [N,64,2]
                 const float* __restrict__ Wi,      // [2,512]
                 const float* __restrict__ bl,      // [512]
                 const float* __restrict__ Wp,      // [128,2]
                 const float* __restrict__ bp,      // [2]
                 const _Float16* __restrict__ WhT,  // [512,128] = Wh^T f16
                 _Float16* __restrict__ conc,       // [N,14336]
                 float* __restrict__ out0)          // preds [N,48,2]
{
    __shared__ __align__(16) _Float16 hA[16 * ULSTM];   // current h, A-matrix tile
    __shared__ float zs[16 * ZW];                       // 32 KB pre-activation
    __shared__ float cs[16 * ULSTM];                    // cell state
    __shared__ float WiS[2 * ZW];
    __shared__ float bS[ZW];
    __shared__ float WpS[ULSTM * 2];
    __shared__ float predS[16 * 2];
    __shared__ float redS[256];

    const int tid  = threadIdx.x;
    const int wave = tid >> 5;
    const int lane = tid & 31;
    const int l15  = lane & 15;
    const int n0   = blockIdx.x * 16;

    for (int i = tid; i < 2 * ZW; i += 256) WiS[i] = Wi[i];
    for (int i = tid; i < ZW; i += 256)     bS[i]  = bl[i];
    for (int i = tid; i < ULSTM * 2; i += 256) WpS[i] = Wp[i];
    for (int i = tid; i < 16 * ULSTM; i += 256) { hA[i] = (_Float16)0.0f; cs[i] = 0.0f; }
    __syncthreads();

    for (int t = 0; t < TT; ++t) {
        // ---- z = h @ Wh  (WMMA) ----
        v8f acc[4];
#pragma unroll
        for (int j = 0; j < 4; ++j)
#pragma unroll
            for (int r = 0; r < 8; ++r) acc[j][r] = 0.0f;

#pragma unroll
        for (int kb = 0; kb < 4; ++kb) {
            v16h a = frag_ld(hA, l15, ULSTM, kb * 32, lane);
#pragma unroll
            for (int j = 0; j < 4; ++j) {
                const int nt = wave * 64 + j * 16;
                v16h bfr = frag_ld(WhT, nt + l15, ULSTM, kb * 32, lane);
                acc[j] = __builtin_amdgcn_wmma_f32_16x16x32_f16(
                    false, a, false, bfr, (short)0, acc[j], false, false);
            }
        }
#pragma unroll
        for (int j = 0; j < 4; ++j) {
            const int n = wave * 64 + j * 16 + l15;
            const int mb = (lane & 16) ? 8 : 0;
#pragma unroll
            for (int r = 0; r < 8; ++r) zs[(mb + r) * ZW + n] = acc[j][r];
        }
        __syncthreads();

        // ---- gates (fold in x@Wi + b), update c/h ----
        for (int idx = tid; idx < 16 * ULSTM; idx += 256) {
            const int row = idx >> 7, u = idx & 127;
            float x0, x1;
            if (t < WUL) {
                const float* xp = x + ((size_t)(n0 + row) * TT + t) * 2;
                x0 = xp[0]; x1 = xp[1];
            } else {
                x0 = predS[row * 2 + 0]; x1 = predS[row * 2 + 1];
            }
            const float zi = zs[row*ZW + u      ] + x0*WiS[u      ] + x1*WiS[ZW + u      ] + bS[u      ];
            const float zf = zs[row*ZW + 128 + u] + x0*WiS[128 + u] + x1*WiS[ZW + 128 + u] + bS[128 + u];
            const float zg = zs[row*ZW + 256 + u] + x0*WiS[256 + u] + x1*WiS[ZW + 256 + u] + bS[256 + u];
            const float zo = zs[row*ZW + 384 + u] + x0*WiS[384 + u] + x1*WiS[ZW + 384 + u] + bS[384 + u];
            const float cn = sigf(zf) * cs[idx] + sigf(zi) * tanhf(zg);
            const float hn = sigf(zo) * tanhf(cn);
            cs[idx] = cn;
            hA[idx] = (_Float16)hn;
            if (t < WUL)   // flat_lstm part of concat
                conc[(size_t)(n0 + row) * CONCAT_K + t * ULSTM + u] = (_Float16)hn;
        }
        __syncthreads();

        // ---- pred = tanh(h @ Wp + bp), parallel 8-way partials + LDS tree ---
        {
            const int pair = tid >> 3, part = tid & 7;   // 32 (row,o) pairs
            const int row = pair >> 1, o = pair & 1;
            float s = 0.0f;
#pragma unroll
            for (int uu = 0; uu < 16; ++uu) {
                const int u = part * 16 + uu;
                s += (float)hA[row * ULSTM + u] * WpS[u * 2 + o];
            }
            redS[tid] = s;
        }
        __syncthreads();
        if (tid < 32) {
            const int row = tid >> 1, o = tid & 1;
            float s = bp[o];
#pragma unroll
            for (int p = 0; p < 8; ++p) s += redS[tid * 8 + p];
            const float pv = tanhf(s);
            predS[row * 2 + o] = pv;
            if (t >= WUL)
                out0[(size_t)(n0 + row) * (TRL * 2) + (t - WUL) * 2 + o] = pv;
        }
        __syncthreads();
    }
}

// ---------------- t1 (elementwise, K=2) then t2 = tanh(t1@W2+b2) via WMMA ----
// M-tile = 64 rows: 4 A sub-tiles reuse each W2T B fragment (4x less L2 traffic)
__global__ __launch_bounds__(256)
void t1t2_kernel(const float* __restrict__ preds,   // [N*48, 2] (== out slice 0)
                 const float* __restrict__ W1,      // [2,256]
                 const float* __restrict__ b1,      // [256]
                 const _Float16* __restrict__ W2T,  // [256,256] = W2^T f16
                 const float* __restrict__ b2,      // [256]
                 _Float16* __restrict__ conc)       // [N,14336]
{
    __shared__ __align__(16) _Float16 t1s[64 * FCN];  // 32 KB
    const int tid = threadIdx.x, wave = tid >> 5, lane = tid & 31, l15 = lane & 15;
    const int m0 = blockIdx.x * 64;

    for (int idx = tid; idx < 64 * FCN; idx += 256) {
        const int row = idx >> 8, j = idx & 255;
        const float p0 = preds[(size_t)(m0 + row) * 2 + 0];
        const float p1 = preds[(size_t)(m0 + row) * 2 + 1];
        t1s[idx] = (_Float16)tanhf(p0 * W1[j] + p1 * W1[FCN + j] + b1[j]);
    }
    __syncthreads();

    v8f acc[4][2];
#pragma unroll
    for (int ms = 0; ms < 4; ++ms)
#pragma unroll
        for (int j = 0; j < 2; ++j)
#pragma unroll
            for (int r = 0; r < 8; ++r) acc[ms][j][r] = 0.0f;

#pragma unroll
    for (int kb = 0; kb < 8; ++kb) {
        v16h b0 = frag_ld(W2T, wave * 32 + l15,      FCN, kb * 32, lane);
        v16h b1f = frag_ld(W2T, wave * 32 + 16 + l15, FCN, kb * 32, lane);
#pragma unroll
        for (int ms = 0; ms < 4; ++ms) {
            v16h a = frag_ld(t1s, ms * 16 + l15, FCN, kb * 32, lane);
            acc[ms][0] = __builtin_amdgcn_wmma_f32_16x16x32_f16(
                false, a, false, b0, (short)0, acc[ms][0], false, false);
            acc[ms][1] = __builtin_amdgcn_wmma_f32_16x16x32_f16(
                false, a, false, b1f, (short)0, acc[ms][1], false, false);
        }
    }

#pragma unroll
    for (int ms = 0; ms < 4; ++ms)
#pragma unroll
        for (int j = 0; j < 2; ++j) {
            const int col = wave * 32 + j * 16 + l15;
            const int mb  = (lane & 16) ? 8 : 0;
            const float bb = b2[col];
#pragma unroll
            for (int r = 0; r < 8; ++r) {
                const int m = m0 + ms * 16 + mb + r;
                const int n = m / TRL, tr = m - n * TRL;
                conc[(size_t)n * CONCAT_K + WUL * ULSTM + tr * FCN + col] =
                    (_Float16)tanhf(acc[ms][j][r] + bb);
            }
        }
}

// ---------------- ev = tanh(concat @ Wc + bc): K=14336 WMMA loop -------------
// M-tile = 32; A tile double-buffered in LDS via async global->LDS DMA.
#define EV_MT 32
#define EV_KC 64                      // halfs per K chunk (2 x K32 steps)
#define EV_NC (CONCAT_K / EV_KC)      // 224 chunks

__global__ __launch_bounds__(256)
void ev_kernel(const _Float16* __restrict__ conc,  // [4096, 14336]
               const _Float16* __restrict__ WcT,   // [256, 14336] = Wc^T f16
               const float* __restrict__ bc,       // [256]
               float* __restrict__ ev)             // [4096, 256]
{
    __shared__ __align__(16) _Float16 aT[2][EV_MT * EV_KC];  // 2 x 4 KB

    const int tid = threadIdx.x, wave = tid >> 5, lane = tid & 31, l15 = lane & 15;
    const int m0 = blockIdx.x * EV_MT;

    // each thread owns one 16B segment per chunk: 32 rows x 8 segs = 256
    const int crow = tid >> 3, cseg = tid & 7;
    const _Float16* gbase = conc + (size_t)(m0 + crow) * CONCAT_K + cseg * 8;
    const uint32_t lds0 = (uint32_t)(uintptr_t)(const void*)&aT[0][crow * EV_KC + cseg * 8];
    const uint32_t lds1 = (uint32_t)(uintptr_t)(const void*)&aT[1][crow * EV_KC + cseg * 8];

    v8f acc[2][2];
#pragma unroll
    for (int ms = 0; ms < 2; ++ms)
#pragma unroll
        for (int j = 0; j < 2; ++j)
#pragma unroll
            for (int r = 0; r < 8; ++r) acc[ms][j][r] = 0.0f;

    async_cp16(lds0, gbase);                       // prefetch chunk 0
    for (int kb = 0; kb < EV_NC; ++kb) {
        if (kb + 1 < EV_NC) {                      // prefetch next chunk
            async_cp16((kb & 1) ? lds0 : lds1, gbase + (size_t)(kb + 1) * EV_KC);
            asm volatile("s_wait_asynccnt 0x1" ::: "memory");  // chunk kb landed
        } else {
            asm volatile("s_wait_asynccnt 0x0" ::: "memory");
        }
        __syncthreads();

        const _Float16* at = aT[kb & 1];
#pragma unroll
        for (int sub = 0; sub < 2; ++sub) {
            const int kk = kb * EV_KC + sub * 32;
            v16h b0 = frag_ld(WcT, wave * 32 + l15,      CONCAT_K, kk, lane);
            v16h b1 = frag_ld(WcT, wave * 32 + 16 + l15, CONCAT_K, kk, lane);
#pragma unroll
            for (int ms = 0; ms < 2; ++ms) {
                v16h a = frag_ld(at, ms * 16 + l15, EV_KC, sub * 32, lane);
                acc[ms][0] = __builtin_amdgcn_wmma_f32_16x16x32_f16(
                    false, a, false, b0, (short)0, acc[ms][0], false, false);
                acc[ms][1] = __builtin_amdgcn_wmma_f32_16x16x32_f16(
                    false, a, false, b1, (short)0, acc[ms][1], false, false);
            }
        }
        __syncthreads();
    }

#pragma unroll
    for (int ms = 0; ms < 2; ++ms)
#pragma unroll
        for (int j = 0; j < 2; ++j) {
            const int col = wave * 32 + j * 16 + l15;
            const int mb  = (lane & 16) ? 8 : 0;
            const float bb = bc[col];
#pragma unroll
            for (int r = 0; r < 8; ++r)
                ev[(size_t)(m0 + ms * 16 + mb + r) * FCN + col] =
                    tanhf(acc[ms][j][r] + bb);
        }
}

// ---------------- softmax / cost heads --------------------------------------
__global__ void heads_kernel(const float* __restrict__ ev,
                             const float* __restrict__ Wprob, const float* __restrict__ bprob,
                             const float* __restrict__ Wasso, const float* __restrict__ basso,
                             float* __restrict__ out) {
    const int n = blockIdx.x * blockDim.x + threadIdx.x;
    if (n >= NSEQ) return;
    float a0 = bprob[0], a1 = bprob[1], cst = basso[0];
    for (int u = 0; u < FCN; ++u) {
        const float e = ev[(size_t)n * FCN + u];
        a0  += e * Wprob[u * 2 + 0];
        a1  += e * Wprob[u * 2 + 1];
        cst += e * Wasso[u];
    }
    const float m  = fmaxf(a0, a1);
    const float e0 = __expf(a0 - m), e1 = __expf(a1 - m);
    const float inv = 1.0f / (e0 + e1);
    out[(size_t)OSZ     + (size_t)n * (TRL * 2)    ] = cst;        // out[1,n,0,0]
    out[(size_t)2 * OSZ + (size_t)n * (TRL * 2) + 0] = e0 * inv;   // out[2,n,0,:]
    out[(size_t)2 * OSZ + (size_t)n * (TRL * 2) + 1] = e1 * inv;
}

// ---------------- host launcher ---------------------------------------------
extern "C" void kernel_launch(void* const* d_in, const int* in_sizes, int n_in,
                              void* d_out, int out_size, void* d_ws, size_t ws_size,
                              hipStream_t stream) {
    (void)in_sizes; (void)n_in; (void)out_size; (void)ws_size;
    const float* inputs = (const float*)d_in[0];
    const float* Wi     = (const float*)d_in[1];
    const float* Wh     = (const float*)d_in[2];
    const float* bl     = (const float*)d_in[3];
    const float* Wp     = (const float*)d_in[4];
    const float* bp     = (const float*)d_in[5];
    const float* W1     = (const float*)d_in[6];
    const float* b1     = (const float*)d_in[7];
    const float* W2     = (const float*)d_in[8];
    const float* b2     = (const float*)d_in[9];
    const float* Wc     = (const float*)d_in[10];
    const float* bc     = (const float*)d_in[11];
    const float* Wprob  = (const float*)d_in[12];
    const float* bprob  = (const float*)d_in[13];
    const float* Wasso  = (const float*)d_in[14];
    const float* basso  = (const float*)d_in[15];
    float* out = (float*)d_out;

    // workspace layout (bytes)
    char* ws = (char*)d_ws;
    _Float16* WhT  = (_Float16*)(ws);                       // 512*128*2   = 128 KB
    _Float16* W2T  = (_Float16*)(ws + 131072);              // 256*256*2   = 128 KB
    _Float16* WcT  = (_Float16*)(ws + 262144);              // 256*14336*2 = 7 MB
    _Float16* conc = (_Float16*)(ws + 7602176);             // 4096*14336*2 = 112 MB
    float*    ev   = (float*)   (ws + 7602176 + (size_t)NSEQ * CONCAT_K * 2);

    convT_kernel<<<256, 256, 0, stream>>>(Wh, WhT, ULSTM, ZW);
    convT_kernel<<<256, 256, 0, stream>>>(W2, W2T, FCN, FCN);
    convT_kernel<<<2048, 256, 0, stream>>>(Wc, WcT, CONCAT_K, FCN);
    zero_kernel<<<768, 256, 0, stream>>>(out + OSZ, 2 * OSZ);

    lstm_kernel<<<NSEQ / 16, 256, 0, stream>>>(inputs, Wi, bl, Wp, bp, WhT, conc, out);
    t1t2_kernel<<<(NSEQ * TRL) / 64, 256, 0, stream>>>(out, W1, b1, W2T, b2, conc);
    ev_kernel<<<NSEQ / EV_MT, 256, 0, stream>>>(conc, WcT, bc, ev);
    heads_kernel<<<NSEQ / 256, 256, 0, stream>>>(ev, Wprob, bprob, Wasso, basso, out);
    copy3_kernel<<<1536, 256, 0, stream>>>(inputs, out + (size_t)3 * OSZ);
}
